// NylonRNN_42846593745198
// MI455X (gfx1250) — matrix operand order repaired and checked by reading, compiled, not run on hardware
//
#include <hip/hip_runtime.h>

#define DEVFN __device__ __forceinline__

typedef _Float16 v8h  __attribute__((ext_vector_type(8)));
typedef _Float16 v16h __attribute__((ext_vector_type(16)));
typedef float    v8f  __attribute__((ext_vector_type(8)));

namespace {
constexpr int L    = 64;    // layers
constexpr int H    = 50;    // hidden
constexpr int DIN  = 60;    // input feature dim (layer 0)
constexpr int DOUT = 50;    // fc output dim
constexpr int B    = 32;    // batch
constexpr int T    = 2048;  // timesteps
constexpr int G4H  = 4 * H; // 200 gate columns
constexpr int KP   = 64;    // K padded (covers 50 and 60)
constexpr int NP   = 208;   // 4H padded to 13 tiles of 16
constexpr int NTIL = NP / 16;       // 13
constexpr int THREADS = NTIL * 32;  // 416 = 13 wave32s
constexpr int CP   = 52;    // padded c-state row stride
constexpr int CH   = 128;   // pipeline chunk (timesteps)
constexpr int NCHUNK = T / CH;
constexpr size_t CHUNK_HALVES = (size_t)2 * CH * B * KP; // depth-2 ring per layer
constexpr size_t FLAG_BYTES = 1024;
}

// ---- fast transcendentals for the serial cell update (native trans ops) ----
#if __has_builtin(__builtin_amdgcn_rcpf)
DEVFN float frcp(float x) { return __builtin_amdgcn_rcpf(x); }
#else
DEVFN float frcp(float x) { return 1.f / x; }
#endif

DEVFN float sigm(float x) { return frcp(1.f + __expf(-x)); }

#if __has_builtin(__builtin_amdgcn_tanhf)
DEVFN float fast_tanh(float x) { return __builtin_amdgcn_tanhf(x); }
#else
// tanh(x) = 1 - 2/(exp(2x)+1); exp->inf and exp->0 limits give exactly +/-1.
DEVFN float fast_tanh(float x) { return 1.f - 2.f * frcp(1.f + __expf(2.f * x)); }
#endif

// ---- async LDS<->global path (CDNA5), guarded so compile never breaks ----
#if __has_builtin(__builtin_amdgcn_global_load_async_to_lds_b128) && \
    __has_builtin(__builtin_amdgcn_global_store_async_from_lds_b128) && \
    __has_builtin(__builtin_amdgcn_s_wait_asynccnt)
#define USE_ASYNC_LDS 1
#endif

#ifdef USE_ASYNC_LDS
// Exact pointee type from the builtin prototype: gcc-vector int4, with
// AS1 (printed as __device__) for global and AS3 (__shared__) for LDS.
typedef int v4i_vs __attribute__((vector_size(16)));
typedef __attribute__((address_space(1))) v4i_vs GV4;
typedef __attribute__((address_space(3))) v4i_vs LV4;

DEVFN void async_ld128(const _Float16* gsrc, _Float16* ldst) {
  __builtin_amdgcn_global_load_async_to_lds_b128(
      (GV4*)(unsigned long long)gsrc,
      (LV4*)(unsigned)(unsigned long long)ldst,  // low 32 bits == LDS offset
      0, 0);
}
DEVFN void async_st128(const _Float16* lsrc, _Float16* gdst) {
  __builtin_amdgcn_global_store_async_from_lds_b128(
      (GV4*)(unsigned long long)gdst,
      (LV4*)(unsigned)(unsigned long long)lsrc,
      0, 0);
}
DEVFN void async_wait0() { __builtin_amdgcn_s_wait_asynccnt(0); }
#else
DEVFN void async_wait0() {}
#endif

// 16-bit A-fragment (16x32, MxK): per ISA, lane m holds two contiguous 8-half
// runs of K: [k0 + hi*8 .. +7] and [k0 + 16 + hi*8 .. +7]. Two 16B loads.
DEVFN v16h load_fragA(const _Float16* p) {
  v8h lo = *(const v8h*)p;
  v8h hi = *(const v8h*)(p + 16);
  return __builtin_shufflevector(lo, hi, 0,1,2,3,4,5,6,7,8,9,10,11,12,13,14,15);
}

// ---------------------------------------------------------------------------
// Persistent layer-pipelined LSTM. One workgroup per layer; weights resident
// in LDS for all T steps; inter-layer activations handed off through an
// L2-resident chunked double buffer with acquire/release flag counters and
// (when available) the gfx1250 async LDS<->global data path.
// ---------------------------------------------------------------------------
__global__ __launch_bounds__(THREADS, 1)
void lstm_pipeline_kernel(
    const float* __restrict__ x,       // [B, DIN, T]
    const float* __restrict__ hn,      // [L, B, H]
    const float* __restrict__ cn,      // [L, B, H]
    const float* __restrict__ Wih0,    // [4H, DIN]
    const float* __restrict__ Wih,     // [L-1, 4H, H]
    const float* __restrict__ Whh,     // [L, 4H, H]
    const float* __restrict__ bih,     // [L, 4H]
    const float* __restrict__ bhh,     // [L, 4H]
    float* __restrict__ out,           // [B*DOUT*T | L*B*H | L*B*H]
    unsigned* __restrict__ produced,   // [L]
    unsigned* __restrict__ consumed,   // [L]
    _Float16* __restrict__ chunkBase,  // L * CHUNK_HALVES halves
    _Float16* __restrict__ ysF)        // [T, B, KP] halves (layer L-1 output)
{
  __shared__ __align__(32) _Float16 sWih[NP * KP];   // [N][K] f16, 26.6 KB
  __shared__ __align__(32) _Float16 sWhh[NP * KP];   // 26.6 KB
  __shared__ __align__(16) float    sBias[NP];
  __shared__ __align__(32) _Float16 sX[B * KP];      // x_t, K padded
  __shared__ __align__(32) _Float16 sH[B * KP];      // h_{t-1}
  __shared__ float sC[B * CP];                        // c state
  __shared__ float sG[B * NP];                        // gate pre-activations

  const int l    = blockIdx.x;
  const int tid  = threadIdx.x;
  const int lane = tid & 31;
  const int ln   = lane & 15;
  const int hsel = lane >> 4;       // half-wave select
  const int wid  = tid >> 5;        // wave id == N-tile id (0..12)

  // ---- one-time staging: weights (reused 2048x), bias, initial h/c ----
  const float* wihSrc = (l == 0) ? Wih0 : (Wih + (size_t)(l - 1) * G4H * H);
  const int    din    = (l == 0) ? DIN : H;
  const float* whhSrc = Whh + (size_t)l * G4H * H;
  for (int idx = tid; idx < NP * KP; idx += THREADS) {
    const int n = idx / KP, k = idx % KP;
    sWih[idx] = (_Float16)((n < G4H && k < din) ? wihSrc[n * din + k] : 0.f);
    sWhh[idx] = (_Float16)((n < G4H && k < H)   ? whhSrc[n * H   + k] : 0.f);
  }
  for (int idx = tid; idx < NP; idx += THREADS)
    sBias[idx] = (idx < G4H) ? (bih[(size_t)l * G4H + idx] + bhh[(size_t)l * G4H + idx]) : 0.f;
  for (int idx = tid; idx < B * KP; idx += THREADS) {
    const int b = idx / KP, k = idx % KP;
    sH[idx] = (_Float16)((k < H) ? hn[((size_t)l * B + b) * H + k] : 0.f);
    sX[idx] = (_Float16)0.f;  // pad region stays zero for all t
  }
  for (int idx = tid; idx < B * CP; idx += THREADS) {
    const int b = idx / CP, k = idx % CP;
    sC[idx] = (k < H) ? cn[((size_t)l * B + b) * H + k] : 0.f;
  }
  __syncthreads();

  _Float16* myChunk = chunkBase + (size_t)l * CHUNK_HALVES;
  const _Float16* upChunk = (l > 0) ? (chunkBase + (size_t)(l - 1) * CHUNK_HALVES) : chunkBase;

  for (int t = 0; t < T; ++t) {
    const int tc    = t % CH;
    const int chunk = t / CH;

    // ---- pipeline flow control at chunk boundaries ----
    if (tc == 0) {
      if (tid == 0) {
        if (l > 0)  // wait for upstream layer to finish this chunk
          while (__hip_atomic_load(&produced[l - 1], __ATOMIC_ACQUIRE,
                                   __HIP_MEMORY_SCOPE_AGENT) < (unsigned)(chunk + 1))
            __builtin_amdgcn_s_sleep(2);
        if (l < L - 1 && chunk >= 2)  // wait for our ring slot to be freed
          while (__hip_atomic_load(&consumed[l], __ATOMIC_ACQUIRE,
                                   __HIP_MEMORY_SCOPE_AGENT) < (unsigned)(chunk - 1))
            __builtin_amdgcn_s_sleep(2);
      }
      __syncthreads();
    }

    // ---- stage x_t into LDS ----
    if (l == 0) {
      for (int idx = tid; idx < B * DIN; idx += THREADS) {
        const int b = idx / DIN, d = idx % DIN;
        sX[b * KP + d] = (_Float16)x[((size_t)b * DIN + d) * T + t];
      }
    } else {
      const _Float16* src = upChunk + (size_t)(chunk & 1) * CH * B * KP + (size_t)tc * B * KP;
      if (tc == 0 && chunk + 1 < NCHUNK && tid < 128)
        __builtin_prefetch(upChunk + (size_t)((chunk + 1) & 1) * CH * B * KP + (size_t)tid * 32, 0, 1);
#ifdef USE_ASYNC_LDS
      for (int idx = tid; idx < (B * KP) / 8; idx += THREADS)   // 256 x b128
        async_ld128(src + idx * 8, sX + idx * 8);
#else
      for (int idx = tid; idx < B * H; idx += THREADS) {
        const int b = idx / H, n = idx % H;
        sX[b * KP + n] = src[b * KP + n];
      }
#endif
    }
    // completes this wave's async loads AND last step's async h stores
    async_wait0();
    __syncthreads();

    // ---- WMMA: G = x_t @ Wih^T + h_{t-1} @ Whh^T + bias ----
    {
      const float bia = sBias[wid * 16 + ln];   // bias folded into C init
      v8f acc0 = {bia, bia, bia, bia, bia, bia, bia, bia};
      v8f acc1 = acc0;
      const int hi8 = hsel * 8;
#pragma unroll
      for (int kt = 0; kt < 2; ++kt) {
        const int k0 = kt * 32;
        // B fragments (KxN, weights stored [N][K]): two 16B K-runs per lane
        v16h bx = load_fragA(&sWih[(wid * 16 + ln) * KP + k0 + hi8]);
        v16h bh = load_fragA(&sWhh[(wid * 16 + ln) * KP + k0 + hi8]);
        // A fragments for both M-tiles of the batch
        v16h ax0 = load_fragA(&sX[(     ln) * KP + k0 + hi8]);
        v16h ax1 = load_fragA(&sX[(16 + ln) * KP + k0 + hi8]);
        v16h ah0 = load_fragA(&sH[(     ln) * KP + k0 + hi8]);
        v16h ah1 = load_fragA(&sH[(16 + ln) * KP + k0 + hi8]);
        acc0 = __builtin_amdgcn_wmma_f32_16x16x32_f16(false, ax0, false, bx, (short)0, acc0, false, false);
        acc1 = __builtin_amdgcn_wmma_f32_16x16x32_f16(false, ax1, false, bx, (short)0, acc1, false, false);
        acc0 = __builtin_amdgcn_wmma_f32_16x16x32_f16(false, ah0, false, bh, (short)0, acc0, false, false);
        acc1 = __builtin_amdgcn_wmma_f32_16x16x32_f16(false, ah1, false, bh, (short)0, acc1, false, false);
      }
      const int colg = wid * 16 + ln;
#pragma unroll
      for (int r = 0; r < 8; ++r) {  // C layout: VGPR r -> row r (+8 for upper half-wave)
        const int row = r + hsel * 8;
        sG[row * NP + colg]        = acc0[r];
        sG[(16 + row) * NP + colg] = acc1[r];
      }
    }
    __syncthreads();

    // ---- elementwise cell update ----
    {
#ifdef USE_ASYNC_LDS
      const bool directStore = (l == L - 1);   // chunks go out via async path
#else
      const bool directStore = true;
#endif
      _Float16* dst = (l < L - 1)
          ? (myChunk + (size_t)(chunk & 1) * CH * B * KP + (size_t)tc * B * KP)
          : (ysF + (size_t)t * B * KP);
      for (int idx = tid; idx < B * H; idx += THREADS) {
        const int b = idx / H, n = idx % H;
        const float gi = sG[b * NP + n];
        const float gf = sG[b * NP + H     + n];
        const float gg = sG[b * NP + 2 * H + n];
        const float go = sG[b * NP + 3 * H + n];
        const float c  = sigm(gf) * sC[b * CP + n] + sigm(gi) * fast_tanh(gg);
        const float h  = sigm(go) * fast_tanh(c);
        sC[b * CP + n] = c;
        const _Float16 hh = (_Float16)h;
        sH[b * KP + n] = hh;
        if (directStore) dst[b * KP + n] = hh;
      }
    }
    __syncthreads();

#ifdef USE_ASYNC_LDS
    // hand h_t downstream straight out of LDS; overlaps next step's x_t staging.
    if (l < L - 1) {
      _Float16* drow = myChunk + (size_t)(chunk & 1) * CH * B * KP + (size_t)tc * B * KP;
      for (int idx = tid; idx < (B * KP) / 8; idx += THREADS)
        async_st128(sH + idx * 8, drow + idx * 8);
    }
#endif

    if (tc == CH - 1) {
      async_wait0();        // chunk must be fully in memory before publishing
      __threadfence();
      if (tid == 0) {
        if (l < L - 1)
          __hip_atomic_store(&produced[l], (unsigned)(chunk + 1),
                             __ATOMIC_RELEASE, __HIP_MEMORY_SCOPE_AGENT);
        if (l > 0)
          __hip_atomic_store(&consumed[l - 1], (unsigned)(chunk + 1),
                             __ATOMIC_RELEASE, __HIP_MEMORY_SCOPE_AGENT);
      }
    }
  }

  // ---- final per-layer h/c state -> d_out tail ----
  for (int idx = tid; idx < B * H; idx += THREADS) {
    const int b = idx / H, n = idx % H;
    out[(size_t)B * DOUT * T + ((size_t)l * B + b) * H + n]                     = (float)sH[b * KP + n];
    out[(size_t)B * DOUT * T + (size_t)L * B * H + ((size_t)l * B + b) * H + n] = sC[b * CP + n];
  }
}

// ---------------------------------------------------------------------------
// out[b,d,t] = sigmoid(ys[t,b,:] @ fc_w[d,:] + fc_b[d]) via WMMA.
// Rows = T*B = 65536 (M), N = 50 -> 4 tiles, K = 50 -> 64 padded.
// ---------------------------------------------------------------------------
__global__ __launch_bounds__(256, 4)
void fc_sigmoid_kernel(const _Float16* __restrict__ ysF,  // [T, B, KP]
                       const float* __restrict__ fcw,     // [DOUT, H]
                       const float* __restrict__ fcb,     // [DOUT]
                       float* __restrict__ out)           // [B, DOUT, T]
{
  __shared__ __align__(32) _Float16 sW[64 * 64];
  __shared__ float sB[64];
  const int tid = threadIdx.x;
  for (int idx = tid; idx < 64 * 64; idx += 256) {
    const int n = idx / 64, k = idx % 64;
    sW[idx] = (_Float16)((n < DOUT && k < H) ? fcw[n * H + k] : 0.f);
  }
  for (int idx = tid; idx < 64; idx += 256)
    sB[idx] = (idx < DOUT) ? fcb[idx] : 0.f;
  __syncthreads();

  const int lane = tid & 31, ln = lane & 15, hsel = lane >> 4;
  const int wid = tid >> 5;               // 0..7
  const int mt = wid >> 2, nt = wid & 3;  // 2 M-tiles x 4 N-tiles per block
  const long rowBase = (long)blockIdx.x * 32 + mt * 16;

  const float bia = sB[nt * 16 + ln];
  v8f acc = {bia, bia, bia, bia, bia, bia, bia, bia};
#pragma unroll
  for (int kt = 0; kt < 2; ++kt) {
    const int k0 = kt * 32;
    v16h bf = load_fragA(&sW[(nt * 16 + ln) * 64 + k0 + hsel * 8]);
    v16h af = load_fragA(ysF + (rowBase + ln) * KP + k0 + hsel * 8);
    acc = __builtin_amdgcn_wmma_f32_16x16x32_f16(false, af, false, bf, (short)0, acc, false, false);
  }
  const int col = nt * 16 + ln;
  if (col < DOUT) {
#pragma unroll
    for (int r = 0; r < 8; ++r) {
      const long row = rowBase + r + hsel * 8;  // row = t*B + b
      const long tt = row / B, bb = row % B;
      out[((size_t)bb * DOUT + col) * T + tt] = sigm(acc[r]);
    }
  }
}

extern "C" void kernel_launch(void* const* d_in, const int* in_sizes, int n_in,
                              void* d_out, int out_size, void* d_ws, size_t ws_size,
                              hipStream_t stream) {
  (void)in_sizes; (void)n_in; (void)out_size; (void)ws_size;
  const float* x    = (const float*)d_in[0];
  const float* hn   = (const float*)d_in[1];
  const float* cn   = (const float*)d_in[2];
  const float* Wih0 = (const float*)d_in[3];
  const float* Wih  = (const float*)d_in[4];
  const float* Whh  = (const float*)d_in[5];
  const float* bih  = (const float*)d_in[6];
  const float* bhh  = (const float*)d_in[7];
  const float* fcw  = (const float*)d_in[8];
  const float* fcb  = (const float*)d_in[9];
  float* out = (float*)d_out;

  // workspace: [flags 1KB | per-layer depth-2 activation rings | final ys]
  unsigned* produced  = (unsigned*)d_ws;
  unsigned* consumed  = produced + L;
  _Float16* chunkBase = (_Float16*)((char*)d_ws + FLAG_BYTES);
  _Float16* ysF       = chunkBase + (size_t)L * CHUNK_HALVES;

  (void)hipMemsetAsync(d_ws, 0, FLAG_BYTES, stream);  // reset pipeline flags

  lstm_pipeline_kernel<<<L, THREADS, 0, stream>>>(
      x, hn, cn, Wih0, Wih, Whh, bih, bhh, out, produced, consumed, chunkBase, ysF);

  fc_sigmoid_kernel<<<(T * B) / 32, 256, 0, stream>>>(ysF, fcw, fcb, out);
}
